// CustomLSTM_42588895707424
// MI455X (gfx1250) — matrix operand order repaired and checked by reading, not compiled, not run on hardware
//
#include <hip/hip_runtime.h>
#include <hip/hip_bf16.h>

// ---------------------------------------------------------------------------
// Sizes (fixed by the reference)
// ---------------------------------------------------------------------------
#define T_STEPS 1024
#define BATCH   256
#define IN_DIM  21
#define HID     256
#define G4      1024        // 4*HID
#define OUTD    64
#define HH      (BATCH*HID) // one h/c layer slab (elements)
#define NLAYER  6
#define NWG     32
#define TPB     256

typedef __bf16 bf16_t;
typedef bf16_t v8bf  __attribute__((ext_vector_type(8)));
typedef bf16_t v16bf __attribute__((ext_vector_type(16)));
typedef float  v8f   __attribute__((ext_vector_type(8)));

// ---------------------------------------------------------------------------
// WMMA helpers (CDNA5: D = A(16x32 bf16) * B(32x16 bf16) + C(16x16 f32))
// ---------------------------------------------------------------------------
static __device__ inline v8f wmma_bf16(v16bf a, v16bf b, v8f c) {
  // (neg_a, A, neg_b, B, c_mod, C, reuse_a, reuse_b)
  return __builtin_amdgcn_wmma_f32_16x16x32_bf16(false, a, false, b, (short)0, c,
                                                 false, false);
}

// A fragment: row-major activations [rows x ld]; lane&15 = M row,
// lane>>4 selects K half-group; VGPRs 0..3 hold K = kb..kb+7,
// VGPRs 4..7 hold K = kb+16..kb+23 (two contiguous 16B loads per lane).
static __device__ inline v16bf load_a_frag(const bf16_t* __restrict__ A, int ld,
                                           int row0, int k0) {
  const int lane = threadIdx.x & 31;
  const int r    = row0 + (lane & 15);
  const int kb   = k0 + ((lane >> 4) << 3);
  const bf16_t* base = A + (size_t)r * ld + kb;
  v8bf lo = *(const v8bf*)(base);
  v8bf hi = *(const v8bf*)(base + 16);
  v16bf out;
#pragma unroll
  for (int i = 0; i < 8; ++i) { out[i] = lo[i]; out[i + 8] = hi[i]; }
  return out;
}

// B fragment: K-major transposed weights Wt[K][ld]; lane&15 = N column,
// lane>>4 selects K half-group; elements 0..7 hold K = kb..kb+7,
// elements 8..15 hold K = kb+16..kb+23 (same K order as the A fragment).
static __device__ inline v16bf load_b_frag(const bf16_t* __restrict__ Wt, int ld,
                                           int k0, int n0) {
  const int lane = threadIdx.x & 31;
  const int n    = n0 + (lane & 15);
  const int kb   = k0 + ((lane >> 4) << 3);
  v16bf out;
#pragma unroll
  for (int i = 0; i < 8; ++i) {
    out[i]     = Wt[(size_t)(kb + i) * ld + n];
    out[i + 8] = Wt[(size_t)(kb + 16 + i) * ld + n];
  }
  return out;
}

static __device__ inline v8f splat8(float x) {
  v8f v;
#pragma unroll
  for (int i = 0; i < 8; ++i) v[i] = x;
  return v;
}

static __device__ inline float sigm(float x) { return 1.0f / (1.0f + __expf(-x)); }

// Device-wide barrier: monotonically increasing arrival counter in d_ws.
static __device__ inline void grid_barrier(unsigned* bar, unsigned target) {
  __threadfence();          // make this thread's global writes visible
  __syncthreads();
  if (threadIdx.x == 0) {
    __hip_atomic_fetch_add(bar, 1u, __ATOMIC_RELEASE, __HIP_MEMORY_SCOPE_AGENT);
    while (__hip_atomic_load(bar, __ATOMIC_ACQUIRE, __HIP_MEMORY_SCOPE_AGENT) <
           target) {
      __builtin_amdgcn_s_sleep(1);
    }
  }
  __syncthreads();
  __threadfence();          // acquire: discard stale cached lines
}

// ---------------------------------------------------------------------------
// Setup kernels
// ---------------------------------------------------------------------------
__global__ void k_xpad(const float* __restrict__ x, bf16_t* __restrict__ Xp, int n) {
  int i = blockIdx.x * blockDim.x + threadIdx.x;
  if (i >= n) return;
  int row = i >> 5, kk = i & 31; // pad K: 21 -> 32
  Xp[i] = (kk < IN_DIM) ? (bf16_t)x[(size_t)row * IN_DIM + kk] : (bf16_t)0.0f;
}

__global__ void k_wih0(const float* __restrict__ W, bf16_t* __restrict__ Wt) {
  int i = blockIdx.x * blockDim.x + threadIdx.x; // 32 x 1024, K-major
  if (i >= 32 * G4) return;
  int k = i / G4, n = i % G4;
  Wt[i] = (k < IN_DIM) ? (bf16_t)W[(size_t)n * IN_DIM + k] : (bf16_t)0.0f;
}

// src[N][K] (row-major f32) -> dst[K][N] (K-major bf16)
__global__ void k_transpose(const float* __restrict__ src, bf16_t* __restrict__ dst,
                            int N, int K) {
  int i = blockIdx.x * blockDim.x + threadIdx.x;
  if (i >= N * K) return;
  int k = i / N, n = i % N;
  dst[i] = (bf16_t)src[(size_t)n * K + k];
}

// Fold head: Wc = W2*W1 (stored K-major bf16 [HID][OUTD]), bc = W2*b1 + b2
__global__ void k_fold(const float* __restrict__ W1, const float* __restrict__ b1,
                       const float* __restrict__ W2, const float* __restrict__ b2,
                       bf16_t* __restrict__ WcT, float* __restrict__ bc) {
  int i = blockIdx.x * blockDim.x + threadIdx.x; // HID*OUTD
  if (i >= HID * OUTD) return;
  int k = i / OUTD, n = i % OUTD;
  float s = 0.f;
  for (int m = 0; m < HID; ++m)
    s += W2[(size_t)n * HID + m] * W1[(size_t)m * HID + k];
  WcT[i] = (bf16_t)s;
  if (k == 0) {
    float t = b2[n];
    for (int m = 0; m < HID; ++m) t += W2[(size_t)n * HID + m] * b1[m];
    bc[n] = t;
  }
}

// ---------------------------------------------------------------------------
// Persistent recurrent kernel
// ---------------------------------------------------------------------------
static __device__ inline void out_proj(const bf16_t* __restrict__ h5,
                                       const bf16_t* __restrict__ WcT,
                                       const float* __restrict__ bc,
                                       float* __restrict__ out, int tIdx, int w) {
  const int lane = threadIdx.x & 31;
  const int r = lane & 15, hf = lane >> 4;
  const int mo = w >> 2, no = w & 3; // 16 x 4 tiles of [256 x 64]
  v8f acc = splat8(bc[no * 16 + r]);
#pragma unroll
  for (int k = 0; k < HID; k += 32) {
    v16bf a  = load_a_frag(h5, HID, mo * 16, k);
    v16bf bf = load_b_frag(WcT, OUTD, k, no * 16);
    acc = wmma_bf16(a, bf, acc);
  }
#pragma unroll
  for (int v = 0; v < 8; ++v) {
    int b = mo * 16 + v + (hf << 3);
    out[(size_t)b * (T_STEPS * OUTD) + (size_t)tIdx * OUTD + no * 16 + r] = acc[v];
  }
}

__global__ void __launch_bounds__(TPB, 1)
lstm_main(const bf16_t* __restrict__ Xp, const bf16_t* __restrict__ Wih0T,
          const bf16_t* __restrict__ WihT, const bf16_t* __restrict__ WhhT,
          const float* __restrict__ b0, const float* __restrict__ b_rest,
          const bf16_t* __restrict__ WcT, const float* __restrict__ bc,
          bf16_t* __restrict__ hbuf,   // [2][NLAYER][BATCH][HID], ping-pong by t
          float* __restrict__ cbuf, unsigned* __restrict__ bar,
          float* __restrict__ out) {
  const int lane = threadIdx.x & 31;
  const int w  = blockIdx.x * (TPB / 32) + (threadIdx.x >> 5); // 0..255
  const int mt = w >> 4, nt = w & 15;                          // 16x16 tile grid
  const int r = lane & 15, hf = lane >> 4;
  const int j = nt * 16 + r; // hidden column owned by this lane

  unsigned nbar = 0;

  for (int t = 0; t < T_STEPS; ++t) {
    // ping-pong via OFFSETS from the single kernel-arg base pointer, so every
    // access keeps the global address space (global_load_* instead of flat_*)
    const size_t poff = (size_t)(t & 1) * (NLAYER * HH);        // h(t-1) slabs
    const size_t coff = (size_t)((t & 1) ^ 1) * (NLAYER * HH);  // h(t) slabs
    const bf16_t* hprev = hbuf + poff;
    bf16_t*       hcur  = hbuf + coff;

    // Head GEMM for step t-1 (dependency met after last barrier of step t-1);
    // runs concurrently with layer-0 of step t on waves 0..63.
    if (t > 0 && w < 64) out_proj(hprev + 5 * HH, WcT, bc, out, t - 1, w);

    for (int l = 0; l < NLAYER; ++l) {
      const bf16_t* Ain = (l == 0) ? (Xp + (size_t)t * BATCH * 32)
                                   : (hcur + (size_t)(l - 1) * HH);
      const int Kin = (l == 0) ? 32 : HID;
      const bf16_t* Win = (l == 0) ? Wih0T : (WihT + (size_t)(l - 1) * HID * G4);
      const float* bias = (l == 0) ? b0 : (b_rest + (size_t)(l - 1) * G4);

      // four gate accumulators for this (mt,nt) tile, init with bias
      v8f acc0 = splat8(bias[0 * HID + j]);
      v8f acc1 = splat8(bias[1 * HID + j]);
      v8f acc2 = splat8(bias[2 * HID + j]);
      v8f acc3 = splat8(bias[3 * HID + j]);

      // input-projection GEMM: inp @ Wih^T
      for (int k = 0; k < Kin; k += 32) {
        v16bf a = load_a_frag(Ain, Kin, mt * 16, k);
        acc0 = wmma_bf16(a, load_b_frag(Win, G4, k, 0 * HID + nt * 16), acc0);
        acc1 = wmma_bf16(a, load_b_frag(Win, G4, k, 1 * HID + nt * 16), acc1);
        acc2 = wmma_bf16(a, load_b_frag(Win, G4, k, 2 * HID + nt * 16), acc2);
        acc3 = wmma_bf16(a, load_b_frag(Win, G4, k, 3 * HID + nt * 16), acc3);
      }

      // recurrent GEMM: h(t-1) @ Whh^T
      const bf16_t* Ar = hprev + (size_t)l * HH;
      const bf16_t* Wr = WhhT + (size_t)l * HID * G4;
#pragma unroll 2
      for (int k = 0; k < HID; k += 32) {
        v16bf a = load_a_frag(Ar, HID, mt * 16, k);
        acc0 = wmma_bf16(a, load_b_frag(Wr, G4, k, 0 * HID + nt * 16), acc0);
        acc1 = wmma_bf16(a, load_b_frag(Wr, G4, k, 1 * HID + nt * 16), acc1);
        acc2 = wmma_bf16(a, load_b_frag(Wr, G4, k, 2 * HID + nt * 16), acc2);
        acc3 = wmma_bf16(a, load_b_frag(Wr, G4, k, 3 * HID + nt * 16), acc3);
      }

      // fused LSTM cell elementwise (all four gates live in registers)
      float*  crow = cbuf + (size_t)l * HH;
      bf16_t* hrow = hcur + (size_t)l * HH;
#pragma unroll
      for (int v = 0; v < 8; ++v) {
        int b = mt * 16 + v + (hf << 3); // C/D layout: row = v + 8*(lane>>4)
        size_t idx = (size_t)b * HID + j;
        float cp = crow[idx];
        float ig = sigm(acc0[v]);
        float fg = sigm(acc1[v]);
        float gg = tanhf(acc2[v]);
        float og = sigm(acc3[v]);
        float cn = fg * cp + ig * gg;
        crow[idx] = cn;
        hrow[idx] = (bf16_t)(og * tanhf(cn));
      }

      grid_barrier(bar, (++nbar) * (unsigned)NWG);
    }
  }

  // Head GEMM for the final step: h(T-1) lives in the slab written at t=T-1
  if (w < 64) {
    const bf16_t* hlast =
        hbuf + (size_t)(((T_STEPS - 1) & 1) ^ 1) * (NLAYER * HH) + 5 * HH;
    out_proj(hlast, WcT, bc, out, T_STEPS - 1, w);
  }
}

// ---------------------------------------------------------------------------
// Host entry
// ---------------------------------------------------------------------------
extern "C" void kernel_launch(void* const* d_in, const int* in_sizes, int n_in,
                              void* d_out, int out_size, void* d_ws, size_t ws_size,
                              hipStream_t stream) {
  const float* x        = (const float*)d_in[0];
  const float* Wih0     = (const float*)d_in[1];
  const float* Whh0     = (const float*)d_in[2];
  const float* b0       = (const float*)d_in[3];
  const float* Wih_rest = (const float*)d_in[4];
  const float* Whh_rest = (const float*)d_in[5];
  const float* b_rest   = (const float*)d_in[6];
  const float* W1       = (const float*)d_in[7];
  const float* b1       = (const float*)d_in[8];
  const float* W2       = (const float*)d_in[9];
  const float* b2       = (const float*)d_in[10];
  float* out = (float*)d_out;

  char* ws = (char*)d_ws;
  size_t off = 0;
  auto take = [&](size_t bytes) {
    char* p = ws + off;
    off = (off + bytes + 255) & ~(size_t)255;
    return p;
  };
  unsigned* bar   = (unsigned*)take(256);
  bf16_t*   Xp    = (bf16_t*)take((size_t)T_STEPS * BATCH * 32 * 2);
  bf16_t*   Wih0T = (bf16_t*)take((size_t)32 * G4 * 2);
  bf16_t*   WihT  = (bf16_t*)take((size_t)5 * HID * G4 * 2);
  bf16_t*   WhhT  = (bf16_t*)take((size_t)6 * HID * G4 * 2);
  bf16_t*   WcT   = (bf16_t*)take((size_t)HID * OUTD * 2);
  float*    bc    = (float*)take((size_t)OUTD * 4);
  bf16_t*   hbuf  = (bf16_t*)take((size_t)2 * NLAYER * HH * 2);
  float*    cbuf  = (float*)take((size_t)NLAYER * HH * 4);

  hipMemsetAsync(bar, 0, 256, stream);
  hipMemsetAsync(hbuf, 0, (size_t)2 * NLAYER * HH * 2, stream);
  hipMemsetAsync(cbuf, 0, (size_t)NLAYER * HH * 4, stream);

  {
    int n = T_STEPS * BATCH * 32;
    k_xpad<<<(n + 255) / 256, 256, 0, stream>>>(x, Xp, n);
  }
  k_wih0<<<(32 * G4 + 255) / 256, 256, 0, stream>>>(Wih0, Wih0T);
  for (int l = 0; l < 5; ++l)
    k_transpose<<<(HID * G4 + 255) / 256, 256, 0, stream>>>(
        Wih_rest + (size_t)l * G4 * HID, WihT + (size_t)l * HID * G4, G4, HID);
  k_transpose<<<(HID * G4 + 255) / 256, 256, 0, stream>>>(Whh0, WhhT, G4, HID);
  for (int l = 0; l < 5; ++l)
    k_transpose<<<(HID * G4 + 255) / 256, 256, 0, stream>>>(
        Whh_rest + (size_t)l * G4 * HID, WhhT + (size_t)(l + 1) * HID * G4, G4, HID);
  k_fold<<<(HID * OUTD + 255) / 256, 256, 0, stream>>>(W1, b1, W2, b2, WcT, bc);

  lstm_main<<<NWG, TPB, 0, stream>>>(Xp, Wih0T, WihT, WhhT, b0, b_rest, WcT, bc,
                                     hbuf, cbuf, bar, out);
}
